// S5Layer_27367531610162
// MI455X (gfx1250) — compile-verified, matrix operand order checked by reading, not run on hardware
//
#include <hip/hip_runtime.h>

// ---------------------------------------------------------------------------
// S5 layer for MI455X (gfx1250): bf16 WMMA GEMMs + chunked parallel scan.
// B=32, S=2048, D=512, N=64.
// ---------------------------------------------------------------------------

typedef __attribute__((ext_vector_type(16))) __bf16 v16bf;
typedef __attribute__((ext_vector_type(8)))  float  v8f;

#define S5_B   32
#define S5_S   2048
#define S5_D   512
#define S5_N   64
#define S5_BS  (S5_B * S5_S)        // 65536 rows
#define S5_NC  16                   // scan chunks
#define S5_CL  (S5_S / S5_NC)       // 128 steps per chunk
#define DT_MIN_F 0.001f
#define DT_MAX_F 0.1f
#define LN_EPS_F 1e-5f

__device__ __forceinline__ unsigned short f32_to_bf16(float f) {
  unsigned int u = __float_as_uint(f);
  unsigned int r = u + 0x7FFFu + ((u >> 16) & 1u);   // round-to-nearest-even
  return (unsigned short)(r >> 16);
}
__device__ __forceinline__ float bf16_to_f32(unsigned short h) {
  return __uint_as_float(((unsigned int)h) << 16);
}
// K index held by packed dword v (pair base) for half 'hi' of a 16-bit WMMA
// operand (CDNA5 ISA 7.12.2, 16-bit A/B 16x32 layout).
__device__ __forceinline__ int kmap(int v, int hi) {
  return (v < 4) ? (hi * 8 + 2 * v) : (16 + hi * 8 + 2 * (v - 4));
}
__device__ __forceinline__ float clip10(float v) {
  return fminf(fmaxf(v, -10.f), 10.f);
}

// ---------------------------------------------------------------------------
// Kernel 0: pack Bm (N=64 x D=512) and Cm (D=512 x N=64) into bf16 WMMA
// B-fragment layout.
// Bm fragment id: (nt in 0..3, kc in 0..15)  -> B tile [K=d(32), N=n(16)]
// Cm fragment id: (nt in 0..31, kc in 0..1)  -> B tile [K=n(32), N=d(16)]
// Each fragment = 32 lanes x 8 dwords, contiguous per lane.
// ---------------------------------------------------------------------------
__global__ __launch_bounds__(256) void pack_weights(
    const float* __restrict__ Bm, const float* __restrict__ Cm,
    unsigned int* __restrict__ Bf, unsigned int* __restrict__ Cf) {
  int t = blockIdx.x * blockDim.x + threadIdx.x;   // 0..32767
  int half = t >> 14;                               // 0 = Bm, 1 = Cm
  int i = t & 16383;
  int frag = i >> 8;
  int within = i & 255;
  int lane = within >> 3;
  int dw = within & 7;
  int hi = lane >> 4;
  if (half == 0) {
    int nt = frag >> 4, kc = frag & 15;
    int n = nt * 16 + (lane & 15);
    int k = kc * 32 + kmap(dw, hi);                 // k = d index
    unsigned int lo = f32_to_bf16(Bm[n * S5_D + k]);
    unsigned int hv = f32_to_bf16(Bm[n * S5_D + k + 1]);
    Bf[(frag * 32 + lane) * 8 + dw] = lo | (hv << 16);
  } else {
    int nt = frag >> 1, kc = frag & 1;
    int d = nt * 16 + (lane & 15);
    int k = kc * 32 + kmap(dw, hi);                 // k = n index
    unsigned int lo = f32_to_bf16(Cm[d * S5_N + k]);
    unsigned int hv = f32_to_bf16(Cm[d * S5_N + k + 1]);
    Cf[(frag * 32 + lane) * 8 + dw] = lo | (hv << 16);
  }
}

// ---------------------------------------------------------------------------
// Kernel 1: per 16-row tile: clip x -> bf16 LDS, compute dt, then
// Bu = dt * (x @ Bm^T) via v_wmma_f32_16x16x32_bf16, and dA = exp(clip(dt*A)).
// Block = 128 threads (4 waves); wave w owns N-tile w (N=64 -> 4 tiles).
// ---------------------------------------------------------------------------
__global__ __launch_bounds__(128) void s5_gemm1(
    const float* __restrict__ x, const float* __restrict__ tiv,
    const float* __restrict__ A_log, const unsigned int* __restrict__ Bf,
    const float* __restrict__ dt_w, const float* __restrict__ dt_b,
    float* __restrict__ Bu, float* __restrict__ dA) {
  __shared__ unsigned int xb[16 * 256];   // 16 rows x 512 bf16 (packed dwords)
  __shared__ float dts[16];
  const int tid = threadIdx.x;
  const int wave = tid >> 5, lane = tid & 31, hi = lane >> 4;
  const int rowBase = blockIdx.x * 16;

  // Phase 1: cooperative load + clip + bf16 convert.
  const float2* x2 = (const float2*)(x + (size_t)rowBase * S5_D);
  for (int i = tid; i < 16 * 256; i += 128) {
    float2 p = x2[i];
    unsigned int lo = f32_to_bf16(clip10(p.x));
    unsigned int hv = f32_to_bf16(clip10(p.y));
    xb[i] = lo | (hv << 16);
  }
  __syncthreads();

  // Phase 2: dt per row (wave w handles rows 4w..4w+3).
  for (int r = wave * 4; r < wave * 4 + 4; ++r) {
    float acc = 0.f;
    for (int cp = lane; cp < 256; cp += 32) {
      unsigned int u = xb[r * 256 + cp];
      acc += bf16_to_f32((unsigned short)(u & 0xFFFFu)) * dt_w[cp * 2];
      acc += bf16_to_f32((unsigned short)(u >> 16)) * dt_w[cp * 2 + 1];
    }
    for (int m = 16; m; m >>= 1) acc += __shfl_xor(acc, m, 32);
    if (lane == 0) {
      float z = acc + dt_b[0];
      float sg = 1.f / (1.f + __expf(-z));
      float dtv = sg * (DT_MAX_F - DT_MIN_F) + DT_MIN_F;
      dtv = fminf(fmaxf(dtv, 1e-6f), 1.f);
      int rowG = rowBase + r;
      int b = rowG >> 11;             // / 2048
      int s = rowG & 2047;
      int sc = (s < S5_S - 1) ? s : (S5_S - 2);
      dtv *= tiv[b * (S5_S - 1) + sc];
      dts[r] = dtv;
    }
  }
  __syncthreads();

  // Phase 3: WMMA K-loop (K = 512 -> 16 chunks of 32).
  v8f c = {};
  const uint4* bfrag = (const uint4*)(Bf + (((size_t)wave * 16) * 32 + lane) * 8);
  for (int kc = 0; kc < 16; ++kc) {
    union { v16bf v; unsigned int u[8]; } Af, Bv;
    int r = lane & 15;
#pragma unroll
    for (int v = 0; v < 8; ++v) {
      int k = kc * 32 + kmap(v, hi);        // even -> dword aligned
      Af.u[v] = xb[r * 256 + (k >> 1)];
    }
    uint4 q0 = bfrag[kc * 64 + 0];          // (32 lanes * 8 dw)/4 = 64 uint4/frag
    uint4 q1 = bfrag[kc * 64 + 1];
    Bv.u[0] = q0.x; Bv.u[1] = q0.y; Bv.u[2] = q0.z; Bv.u[3] = q0.w;
    Bv.u[4] = q1.x; Bv.u[5] = q1.y; Bv.u[6] = q1.z; Bv.u[7] = q1.w;
    c = __builtin_amdgcn_wmma_f32_16x16x32_bf16(false, Af.v, false, Bv.v,
                                                (short)0, c, false, false);
  }

  // Epilogue: Bu = dt * acc.
  float* BuRow = Bu + (size_t)rowBase * S5_N;
#pragma unroll
  for (int i = 0; i < 8; ++i) {
    int m = i + 8 * hi;
    int n = wave * 16 + (lane & 15);
    BuRow[m * S5_N + n] = c[i] * dts[m];
  }
  // dA = exp(clip(dt * -exp(A_log), -5, 5)) for 16x64 elements.
  float* dARow = dA + (size_t)rowBase * S5_N;
  for (int i = tid; i < 16 * S5_N; i += 128) {
    int m = i >> 6, n = i & 63;
    float Av = -__expf(A_log[n]);
    float e = fminf(fmaxf(dts[m] * Av, -5.f), 5.f);
    dARow[i] = __expf(e);
  }
  // CDNA5 prefetch of the next tile's x (global_prefetch_b8).
  if (rowBase + 16 < S5_BS) {
    const char* nx = (const char*)(x + ((size_t)rowBase + 16) * S5_D) + tid * 256;
    __builtin_prefetch(nx, 0, 1);
  }
}

// ---------------------------------------------------------------------------
// Kernel 2: chunk-local scan. Block = (b, chunk), 64 lanes = 64 states.
// h_local and cumP (running product of dA within chunk) stored; carries out.
// ---------------------------------------------------------------------------
__global__ __launch_bounds__(64) void s5_scan_local(
    const float* __restrict__ dA, const float* __restrict__ Bu,
    float* __restrict__ hloc, float* __restrict__ cumP,
    float* __restrict__ carryH, float* __restrict__ carryP) {
  int n = threadIdx.x;
  int c = blockIdx.x & (S5_NC - 1);
  int b = blockIdx.x >> 4;
  size_t base = ((size_t)(b * S5_S + c * S5_CL)) * S5_N + n;
  float h = 0.f, P = 1.f;
#pragma unroll 4
  for (int s = 0; s < S5_CL; ++s) {
    size_t idx = base + (size_t)s * S5_N;
    float d = dA[idx], u = Bu[idx];
    h = fmaf(d, h, u);
    P *= d;
    hloc[idx] = h;
    cumP[idx] = P;
  }
  int ci = (b * S5_NC + c) * S5_N + n;
  carryH[ci] = h;
  carryP[ci] = P;
}

// ---------------------------------------------------------------------------
// Kernel 3: fold chunk carries serially (16 steps) per (b, n).
// carryIn[b,c,n] = true h at the step before chunk c starts.
// ---------------------------------------------------------------------------
__global__ __launch_bounds__(256) void s5_scan_combine(
    const float* __restrict__ carryH, const float* __restrict__ carryP,
    float* __restrict__ carryIn) {
  int t = blockIdx.x * blockDim.x + threadIdx.x;
  if (t >= S5_B * S5_N) return;
  int b = t >> 6, n = t & 63;
  float H = 0.f;
#pragma unroll
  for (int c = 0; c < S5_NC; ++c) {
    int ci = (b * S5_NC + c) * S5_N + n;
    carryIn[ci] = H;
    H = fmaf(carryP[ci], H, carryH[ci]);
  }
}

// ---------------------------------------------------------------------------
// Kernel 4: y = hs @ Cm^T + Dv*x, then LayerNorm over D. Scan fix-up fused
// into the A-tile load. Block = 256 threads (8 waves) per 16-row tile;
// wave w covers N-tiles 4w..4w+3 (32 tiles of 16 over D=512), K=64 (2 WMMAs).
// ---------------------------------------------------------------------------
__global__ __launch_bounds__(256) void s5_gemm2_ln(
    const float* __restrict__ hloc, const float* __restrict__ cumP,
    const float* __restrict__ carryIn, const unsigned int* __restrict__ Cf,
    const float* __restrict__ x, const float* __restrict__ Dv,
    const float* __restrict__ ln_g, const float* __restrict__ ln_b,
    float* __restrict__ out) {
  __shared__ unsigned int hsb[16 * 32];   // 16 rows x 64 bf16 (packed dwords)
  __shared__ float yf[16 * S5_D];         // 32 KB
  const int tid = threadIdx.x;
  const int wave = tid >> 5, lane = tid & 31, hi = lane >> 4;
  const int rowBase = blockIdx.x * 16;
  const int b = rowBase >> 11;
  const int chunk = (rowBase & 2047) >> 7;   // same for whole 16-row tile

  // Phase 1: reconstruct h tile (fix-up) -> bf16 LDS.
  {
    const float* hp = hloc + (size_t)rowBase * S5_N;
    const float* pp = cumP + (size_t)rowBase * S5_N;
    const float* cin = carryIn + (b * S5_NC + chunk) * S5_N;
    for (int i = tid; i < 16 * 32; i += 256) {
      int e0 = i * 2;
      int n0 = e0 & 63;
      float h0 = fmaf(pp[e0], cin[n0], hp[e0]);
      float h1 = fmaf(pp[e0 + 1], cin[n0 + 1], hp[e0 + 1]);
      hsb[i] = (unsigned int)f32_to_bf16(h0) |
               ((unsigned int)f32_to_bf16(h1) << 16);
    }
  }
  __syncthreads();

  // Phase 2: WMMA over 4 N-tiles per wave.
  for (int tt = 0; tt < 4; ++tt) {
    int nt = wave * 4 + tt;
    v8f c = {};
#pragma unroll
    for (int kc = 0; kc < 2; ++kc) {
      union { v16bf v; unsigned int u[8]; } Af, Bv;
      int r = lane & 15;
#pragma unroll
      for (int v = 0; v < 8; ++v) {
        int k = kc * 32 + kmap(v, hi);
        Af.u[v] = hsb[r * 32 + (k >> 1)];
      }
      const uint4* bp = (const uint4*)(Cf + (((size_t)(nt * 2 + kc)) * 32 + lane) * 8);
      uint4 q0 = bp[0], q1 = bp[1];
      Bv.u[0] = q0.x; Bv.u[1] = q0.y; Bv.u[2] = q0.z; Bv.u[3] = q0.w;
      Bv.u[4] = q1.x; Bv.u[5] = q1.y; Bv.u[6] = q1.z; Bv.u[7] = q1.w;
      c = __builtin_amdgcn_wmma_f32_16x16x32_bf16(false, Af.v, false, Bv.v,
                                                  (short)0, c, false, false);
    }
    // Epilogue: + Dv * clip(x).
#pragma unroll
    for (int i = 0; i < 8; ++i) {
      int m = i + 8 * hi;
      int d = nt * 16 + (lane & 15);
      float xv = clip10(x[((size_t)rowBase + m) * S5_D + d]);
      yf[m * S5_D + d] = c[i] + Dv[d] * xv;
    }
  }
  __syncthreads();

  // Phase 3: LayerNorm, wave handles 2 rows.
  for (int rr = 0; rr < 2; ++rr) {
    int m = wave * 2 + rr;
    float sum = 0.f, sq = 0.f;
    for (int d = lane; d < S5_D; d += 32) {
      float v = yf[m * S5_D + d];
      sum += v; sq += v * v;
    }
    for (int msk = 16; msk; msk >>= 1) {
      sum += __shfl_xor(sum, msk, 32);
      sq  += __shfl_xor(sq,  msk, 32);
    }
    float mu = sum * (1.f / S5_D);
    float var = sq * (1.f / S5_D) - mu * mu;
    float rstd = rsqrtf(var + LN_EPS_F);
    float* orow = out + ((size_t)rowBase + m) * S5_D;
    for (int d = lane; d < S5_D; d += 32) {
      orow[d] = ln_g[d] * (yf[m * S5_D + d] - mu) * rstd + ln_b[d];
    }
  }
}

// ---------------------------------------------------------------------------
// Host launcher.
// ---------------------------------------------------------------------------
extern "C" void kernel_launch(void* const* d_in, const int* in_sizes, int n_in,
                              void* d_out, int out_size, void* d_ws, size_t ws_size,
                              hipStream_t stream) {
  const float* x     = (const float*)d_in[0];
  const float* tiv   = (const float*)d_in[1];
  const float* A_log = (const float*)d_in[2];
  const float* Bm    = (const float*)d_in[3];
  const float* Cm    = (const float*)d_in[4];
  const float* Dv    = (const float*)d_in[5];
  const float* dt_w  = (const float*)d_in[6];
  const float* dt_b  = (const float*)d_in[7];
  const float* ln_g  = (const float*)d_in[8];
  const float* ln_b  = (const float*)d_in[9];
  float* out = (float*)d_out;

  char* ws = (char*)d_ws;
  size_t off = 0;
  auto alloc = [&](size_t bytes) -> void* {
    void* p = ws + off;
    off += (bytes + 255) & ~(size_t)255;
    return p;
  };
  const size_t BSN = (size_t)S5_BS * S5_N;
  float* Bu      = (float*)alloc(BSN * 4);                    // 16 MB
  float* dA      = (float*)alloc(BSN * 4);                    // 16 MB
  float* hloc    = (float*)alloc(BSN * 4);                    // 16 MB
  float* cumP    = (float*)alloc(BSN * 4);                    // 16 MB
  float* carryH  = (float*)alloc((size_t)S5_B * S5_NC * S5_N * 4);
  float* carryP  = (float*)alloc((size_t)S5_B * S5_NC * S5_N * 4);
  float* carryIn = (float*)alloc((size_t)S5_B * S5_NC * S5_N * 4);
  unsigned int* Bf = (unsigned int*)alloc(16384 * 4);         // packed Bm frags
  unsigned int* Cf = (unsigned int*)alloc(16384 * 4);         // packed Cm frags
  (void)ws_size; (void)in_sizes; (void)n_in; (void)out_size;

  pack_weights<<<128, 256, 0, stream>>>(Bm, Cm, Bf, Cf);
  s5_gemm1<<<S5_BS / 16, 128, 0, stream>>>(x, tiv, A_log, Bf, dt_w, dt_b, Bu, dA);
  s5_scan_local<<<S5_B * S5_NC, 64, 0, stream>>>(dA, Bu, hloc, cumP, carryH, carryP);
  s5_scan_combine<<<(S5_B * S5_N + 255) / 256, 256, 0, stream>>>(carryH, carryP, carryIn);
  s5_gemm2_ln<<<S5_BS / 16, 256, 0, stream>>>(hloc, cumP, carryIn, Cf, x, Dv,
                                              ln_g, ln_b, out);
}
